// MultilayerGRU_75213467287686
// MI455X (gfx1250) — matrix operand order repaired and checked by reading, not compiled, not run on hardware
//
#include <hip/hip_runtime.h>
#include <hip/hip_bf16.h>
#include <stdint.h>

// ---------------- problem constants ----------------
#define B_   32
#define S_   1024
#define H_   1024
#define L_   3
#define MT   2          // M tiles (32/16)
#define KT   32         // K tiles (1024/32)
#define NT   64         // N tiles (1024/16)
#define TILE 512        // bf16 elements per packed 16x32 (A) or 32x16 (B) tile
#define MATSZ (1024*1024)     // bf16 elems per packed weight matrix
#define XSTEP (MT*KT*TILE)    // 32768 bf16 elems per packed [32,1024] activation
#define NWG  48

typedef __bf16 v16bf __attribute__((ext_vector_type(16)));
typedef float  v8f   __attribute__((ext_vector_type(8)));

// ---------------- helpers ----------------
__device__ __forceinline__ unsigned short f2bf(float f) {
  union { float f; unsigned u; } v; v.f = f;
  unsigned u = v.u;
  u += 0x7fffu + ((u >> 16) & 1u);          // round-to-nearest-even
  return (unsigned short)(u >> 16);
}
__device__ __forceinline__ float bf2f(unsigned short h) {
  union { unsigned u; float f; } v; v.u = ((unsigned)h) << 16;
  return v.f;
}
// index of element (m,k) of a [32,1024] matrix stored in WMMA A-tile packed order
// A 16x32 bf16 tile: lane<16 holds row m, K = e(0..7)->0..7, e(8..15)->16..23;
// lane>=16 holds K = 8..15 / 24..31 (ISA 7.12.2).
__device__ __forceinline__ int packedA(int m, int k) {
  int mt = m >> 4, kt = k >> 5, mm = m & 15, kk = k & 31;
  int lane = mm + (((kk >> 3) & 1) << 4);
  int e    = (kk & 7) + ((kk >> 4) << 3);
  return (mt * KT + kt) * TILE + lane * 16 + e;
}
__device__ __forceinline__ float sigm(float x) { return 1.f / (1.f + __expf(-x)); }
__device__ __forceinline__ float tanh_f(float x) {
  x = fminf(15.f, fmaxf(-15.f, x));
  float e = __expf(2.f * x);
  return (e - 1.f) / (e + 1.f);
}

// ---------------- pack weights: fp32 [K,N] row-major -> bf16 B-tiles ----------------
// matrices 0..17: layer-major {Wzx,Wzh,Wrx,Wrh,Wgx,Wgh}; 18: Wout
__global__ void pack_weights_kernel(const float* __restrict__ Wzx, const float* __restrict__ Wzh,
                                    const float* __restrict__ Wrx, const float* __restrict__ Wrh,
                                    const float* __restrict__ Wgx, const float* __restrict__ Wgh,
                                    const float* __restrict__ Wout, unsigned short* __restrict__ Wpk) {
  size_t i = (size_t)blockIdx.x * blockDim.x + threadIdx.x;   // < 19 * 2^20
  int mat = (int)(i >> 20);
  int p   = (int)(i & (MATSZ - 1));
  int tileIdx = p >> 9, within = p & 511;
  int lane = within >> 4, e = within & 15;
  int nt = tileIdx >> 5, kt = tileIdx & 31;                   // tile order: nt*32 + kt
  int n = nt * 16 + (lane & 15);                              // B: lane = column
  int k = kt * 32 + (e & 7) + ((e >> 3) << 4) + ((lane >> 4) & 1) * 8;   // lanes 16..31 hold K=8..15/24..31
  const float* base;
  if (mat < 18) {
    int l = mat / 6, w = mat % 6;
    switch (w) {
      case 0: base = Wzx; break; case 1: base = Wzh; break;
      case 2: base = Wrx; break; case 3: base = Wrh; break;
      case 4: base = Wgx; break; default: base = Wgh; break;
    }
    base += (size_t)l * MATSZ;
  } else {
    base = Wout;
  }
  Wpk[i] = f2bf(base[(size_t)k * H_ + n]);
}

// ---------------- pack x: fp32 [B,S,I] -> per-timestep bf16 A-tiles ----------------
__global__ void pack_x_kernel(const float* __restrict__ x, unsigned short* __restrict__ Xpk) {
  size_t i = (size_t)blockIdx.x * blockDim.x + threadIdx.x;   // < S * XSTEP
  int t = (int)(i >> 15);
  int q = (int)(i & (XSTEP - 1));
  int tileIdx = q >> 9, within = q & 511;
  int lane = within >> 4, e = within & 15;
  int mt = tileIdx >> 5, kt = tileIdx & 31;
  int m  = mt * 16 + (lane & 15);
  int kk = (e & 7) + ((e >> 3) << 4) + ((lane & 16) ? 8 : 0);
  int k  = kt * 32 + kk;
  Xpk[i] = f2bf(x[((size_t)m * S_ + t) * H_ + k]);
}

// ---------------- device-wide sense barrier ----------------
__device__ __forceinline__ void gbar(unsigned* bar, unsigned* lg) {
  __syncthreads();
  if (threadIdx.x == 0) {
    unsigned g = *lg;
    __threadfence();
    unsigned a = __hip_atomic_fetch_add(&bar[0], 1u, __ATOMIC_ACQ_REL, __HIP_MEMORY_SCOPE_AGENT);
    if (a == (unsigned)gridDim.x - 1u) {
      __hip_atomic_store(&bar[0], 0u, __ATOMIC_RELAXED, __HIP_MEMORY_SCOPE_AGENT);
      __hip_atomic_fetch_add(&bar[1], 1u, __ATOMIC_RELEASE, __HIP_MEMORY_SCOPE_AGENT);
    } else {
      while (__hip_atomic_load(&bar[1], __ATOMIC_ACQUIRE, __HIP_MEMORY_SCOPE_AGENT) <= g)
        __builtin_amdgcn_s_sleep(2);
    }
    __threadfence();
    *lg = g + 1;
  }
  __syncthreads();
}

// ---------------- persistent GRU kernel ----------------
__global__ void __launch_bounds__(256, 1)
gru_persistent_kernel(const unsigned short* __restrict__ Wpk,
                      const unsigned short* __restrict__ Xpk,
                      unsigned short* __restrict__ Hpk,    // [L][XSTEP] packed A layout
                      unsigned short* __restrict__ RHpk,   // [XSTEP]
                      float* __restrict__ Zbuf,            // [32][1024]
                      float* __restrict__ Gxbuf,           // [32][1024]
                      const float* __restrict__ bzh, const float* __restrict__ brh,
                      const float* __restrict__ bgh, const float* __restrict__ bout,
                      float* __restrict__ out,             // [B,S,O] then [B,L,H]
                      unsigned* __restrict__ bar) {
  const int lane = threadIdx.x & 31;
  const int wave = blockIdx.x * (blockDim.x >> 5) + (threadIdx.x >> 5);
  const int W    = gridDim.x * (blockDim.x >> 5);      // 384 waves total
  unsigned lgen = 0;

  for (int t = 0; t < S_; ++t) {
    for (int l = 0; l < L_; ++l) {
      const unsigned short* xsrc = (l == 0) ? (Xpk + (size_t)t * XSTEP)
                                            : (Hpk + (size_t)(l - 1) * XSTEP);
      unsigned short* hcur = Hpk + (size_t)l * XSTEP;

      // ---- phase A: z, r, and gx = x @ Wgx (384 tiles) ----
      for (int tile = wave; tile < 3 * MT * NT; tile += W) {
        int gate = tile >> 7;            // 0=z 1=r 2=gx
        int mt   = (tile >> 6) & 1;
        int nt   = tile & 63;
        const unsigned short* Wx = Wpk + (size_t)(l * 6 + gate * 2) * MATSZ
                                       + (size_t)nt * KT * TILE + lane * 16;
        const unsigned short* Wh = Wpk + (size_t)(l * 6 + gate * 2 + 1) * MATSZ
                                       + (size_t)nt * KT * TILE + lane * 16;
        const unsigned short* Ax = xsrc + mt * KT * TILE + lane * 16;
        const unsigned short* Ah = hcur + mt * KT * TILE + lane * 16;
        v8f acc = {};
        if (gate < 2) {
          for (int kt = 0; kt < KT; ++kt) {
            v16bf a  = *(const v16bf*)(Ax + kt * TILE);
            v16bf b  = *(const v16bf*)(Wx + kt * TILE);
            acc = __builtin_amdgcn_wmma_f32_16x16x32_bf16(false, a, false, b, (short)0, acc, false, false);
            v16bf ah = *(const v16bf*)(Ah + kt * TILE);
            v16bf bh = *(const v16bf*)(Wh + kt * TILE);
            acc = __builtin_amdgcn_wmma_f32_16x16x32_bf16(false, ah, false, bh, (short)0, acc, false, false);
          }
        } else {
          for (int kt = 0; kt < KT; ++kt) {
            v16bf a = *(const v16bf*)(Ax + kt * TILE);
            v16bf b = *(const v16bf*)(Wx + kt * TILE);
            acc = __builtin_amdgcn_wmma_f32_16x16x32_bf16(false, a, false, b, (short)0, acc, false, false);
          }
        }
        int col   = nt * 16 + (lane & 15);
        int mbase = mt * 16 + ((lane & 16) ? 8 : 0);   // C layout: VGPR v -> row mbase+v
        if (gate == 0) {
          float bias = bzh[l * H_ + col];
          #pragma unroll
          for (int v = 0; v < 8; ++v)
            Zbuf[(mbase + v) * H_ + col] = sigm(acc[v] + bias);
        } else if (gate == 1) {
          float bias = brh[l * H_ + col];
          #pragma unroll
          for (int v = 0; v < 8; ++v) {
            int m = mbase + v;
            float rv = sigm(acc[v] + bias);
            int pi = packedA(m, col);
            RHpk[pi] = f2bf(rv * bf2f(hcur[pi]));      // (r*h) straight into A layout
          }
        } else {
          #pragma unroll
          for (int v = 0; v < 8; ++v)
            Gxbuf[(mbase + v) * H_ + col] = acc[v];
        }
      }
      gbar(bar, &lgen);

      // ---- phase B: g = tanh(gx + (r*h)@Wgh + b), fused h update (128 tiles) ----
      for (int tile = wave; tile < MT * NT; tile += W) {
        int mt = tile >> 6, nt = tile & 63;
        const unsigned short* Wg = Wpk + (size_t)(l * 6 + 5) * MATSZ
                                       + (size_t)nt * KT * TILE + lane * 16;
        const unsigned short* Ar = RHpk + mt * KT * TILE + lane * 16;
        v8f acc = {};
        for (int kt = 0; kt < KT; ++kt) {
          v16bf a = *(const v16bf*)(Ar + kt * TILE);
          v16bf b = *(const v16bf*)(Wg + kt * TILE);
          acc = __builtin_amdgcn_wmma_f32_16x16x32_bf16(false, a, false, b, (short)0, acc, false, false);
        }
        int col   = nt * 16 + (lane & 15);
        int mbase = mt * 16 + ((lane & 16) ? 8 : 0);
        float bias = bgh[l * H_ + col];
        #pragma unroll
        for (int v = 0; v < 8; ++v) {
          int m = mbase + v;
          float g  = tanh_f(acc[v] + Gxbuf[m * H_ + col] + bias);
          float zv = Zbuf[m * H_ + col];
          int pi = packedA(m, col);
          float hold = bf2f(hcur[pi]);
          float hnew = zv * hold + (1.f - zv) * g;
          hcur[pi] = f2bf(hnew);                       // becomes next layer's A operand
          if (t == S_ - 1)
            out[(size_t)B_ * S_ * H_ + ((size_t)m * L_ + l) * H_ + col] = hnew;
        }
      }
      gbar(bar, &lgen);
    }

    // ---- phase D: y_t = h2 @ Wout + bout (128 tiles) ----
    for (int tile = wave; tile < MT * NT; tile += W) {
      int mt = tile >> 6, nt = tile & 63;
      const unsigned short* Wo = Wpk + (size_t)18 * MATSZ + (size_t)nt * KT * TILE + lane * 16;
      const unsigned short* Ah = Hpk + (size_t)2 * XSTEP + mt * KT * TILE + lane * 16;
      v8f acc = {};
      for (int kt = 0; kt < KT; ++kt) {
        v16bf a = *(const v16bf*)(Ah + kt * TILE);
        v16bf b = *(const v16bf*)(Wo + kt * TILE);
        acc = __builtin_amdgcn_wmma_f32_16x16x32_bf16(false, a, false, b, (short)0, acc, false, false);
      }
      int col   = nt * 16 + (lane & 15);
      int mbase = mt * 16 + ((lane & 16) ? 8 : 0);
      float bias = bout[col];
      #pragma unroll
      for (int v = 0; v < 8; ++v) {
        int m = mbase + v;
        out[((size_t)m * S_ + t) * H_ + col] = acc[v] + bias;
      }
    }
    gbar(bar, &lgen);
  }
}

// ---------------- host launcher ----------------
extern "C" void kernel_launch(void* const* d_in, const int* in_sizes, int n_in,
                              void* d_out, int out_size, void* d_ws, size_t ws_size,
                              hipStream_t stream) {
  const float* x    = (const float*)d_in[0];
  const float* Wzx  = (const float*)d_in[1];
  const float* Wzh  = (const float*)d_in[2];
  const float* bzh  = (const float*)d_in[3];
  const float* Wrx  = (const float*)d_in[4];
  const float* Wrh  = (const float*)d_in[5];
  const float* brh  = (const float*)d_in[6];
  const float* Wgx  = (const float*)d_in[7];
  const float* Wgh  = (const float*)d_in[8];
  const float* bgh  = (const float*)d_in[9];
  const float* Wout = (const float*)d_in[10];
  const float* bout = (const float*)d_in[11];
  float* out = (float*)d_out;
  char* ws = (char*)d_ws;

  unsigned short* Wpk  = (unsigned short*)(ws);                        // 19 MB matrices -> 38 MB
  unsigned short* Xpk  = (unsigned short*)(ws + (size_t)40  * 1048576); // 64 MB
  unsigned short* Hpk  = (unsigned short*)(ws + (size_t)104 * 1048576); // 192 KB
  unsigned short* RHpk = (unsigned short*)(ws + (size_t)105 * 1048576); // 64 KB
  float*          Zbuf = (float*)         (ws + (size_t)106 * 1048576); // 128 KB
  float*          Gxbf = (float*)         (ws + (size_t)107 * 1048576); // 128 KB
  unsigned*       bar  = (unsigned*)      (ws + (size_t)108 * 1048576); // 8 B

  hipMemsetAsync(Hpk, 0, (size_t)L_ * XSTEP * sizeof(unsigned short), stream);
  hipMemsetAsync(bar, 0, 256, stream);

  pack_weights_kernel<<<(19 * MATSZ) / 256, 256, 0, stream>>>(Wzx, Wzh, Wrx, Wrh, Wgx, Wgh, Wout, Wpk);
  pack_x_kernel<<<((size_t)S_ * XSTEP) / 256, 256, 0, stream>>>(x, Xpk);
  gru_persistent_kernel<<<NWG, 256, 0, stream>>>(Wpk, Xpk, Hpk, RHpk, Zbuf, Gxbf,
                                                 bzh, brh, bgh, bout, out, bar);
}
